// LearnedDoubleCrossBabyWithSinEmbedding_73486890435289
// MI455X (gfx1250) — matrix operand order repaired and checked by reading, not compiled, not run on hardware
//
#include <hip/hip_runtime.h>

// Problem constants (match the JAX reference).
#define B_  8
#define S_  256
#define V_  4096
#define D_  128

typedef __attribute__((ext_vector_type(8)))  float          v8f;
typedef __attribute__((ext_vector_type(16))) __bf16         v16bf;
typedef __attribute__((ext_vector_type(8)))  unsigned short v8us;

// ---------------------------------------------------------------------------
// Split-bf16 helpers: v = hi + lo (+O(2^-17)).  hi = bit-truncated bf16 of v
// (v - hi is EXACT in f32), lo = bit-truncated bf16 of the remainder.
// A*B ~= Ah*Bh + Ah*Bl + Al*Bh  (dropped Al*Bl <= 2^-16 relative per term).
// ---------------------------------------------------------------------------
__device__ __forceinline__ void split_bf16(float v, unsigned short& h, unsigned short& l) {
  unsigned u = __float_as_uint(v);
  h = (unsigned short)(u >> 16);
  float hif = __uint_as_float(u & 0xffff0000u);
  float lof = v - hif;                         // exact
  l = (unsigned short)(__float_as_uint(lof) >> 16);
}
__device__ __forceinline__ float bf16_to_f(unsigned short u) {
  return __uint_as_float(((unsigned)u) << 16);
}

// CDNA5 bf16 WMMA: D(16x16 f32) = A(16x32 bf16) * B(32x16 bf16) + C
__device__ __forceinline__ v8f wmma_bf16(v16bf a, v16bf b, v8f c) {
  return __builtin_amdgcn_wmma_f32_16x16x32_bf16(
      false, a, false, b, (short)0, c, false, false);
}

// Async copy: 16 bytes per lane, global -> LDS, tracked by ASYNCcnt.
// VDST = per-lane LDS byte address, VADDR = per-lane 64-bit global address.
__device__ __forceinline__ unsigned lds_off(const void* p) {
  // addrspace(3)->generic casts put the LDS byte offset in the low 32 bits.
  return (unsigned)(unsigned long long)p;
}
__device__ __forceinline__ void async_copy16(unsigned lds_byte, const void* g) {
  asm volatile("global_load_async_to_lds_b128 %0, %1, off"
               :: "v"(lds_byte), "v"((unsigned long long)g) : "memory");
}
__device__ __forceinline__ void wait_async0() {
  asm volatile("s_wait_asynccnt 0" ::: "memory");
}

// A fragment (16x32 bf16, MxK) from an LDS tile with ld=32: lane holds row
// m=lane&15; elems 0..7 = K[8*half,+8), elems 8..15 = K[16+8*half,+8).
__device__ __forceinline__ v16bf fragA_lds(const unsigned short* tile, int row, int half) {
  union { v16bf b; v8us u[2]; } f;
  const unsigned short* p = tile + row * 32 + half * 8;
  f.u[0] = *(const v8us*)(p);
  f.u[1] = *(const v8us*)(p + 16);
  return f.b;
}
// B fragment (32x16 bf16, KxN) = W^T from an LDS tile [N,32] K-major:
// lane holds col n=lane&15; elems 0..7 = K[8*half,+8), elems 8..15 = K[16+8*half,+8).
__device__ __forceinline__ v16bf fragB_lds(const unsigned short* tile, int row, int half) {
  union { v16bf b; v8us u[2]; } f;
  const unsigned short* p = tile + row * 32 + half * 8;
  f.u[0] = *(const v8us*)(p);
  f.u[1] = *(const v8us*)(p + 16);
  return f.b;
}

// Sinusoidal positional encoding: ang = s*10000^(-2*(j/2)/dim).
__device__ __forceinline__ float pe_val(int s, int j, int dim) {
  const float LOG2_10000 = 13.287712379549449f;
  float e   = -LOG2_10000 * (2.0f * (float)(j >> 1)) / (float)dim;
  float ang = (float)s * exp2f(e);
  return (j & 1) ? cosf(ang) : sinf(ang);
}

// ---------------------------------------------------------------------------
// Split-bf16 GEMM with double-buffered async-to-LDS staging.
//   C[b] = relu(A[b] * W[b]^T + bias) [+ PE]
// A,W are (hi,lo) bf16 pairs, [rows,K] K-major.  M = S_ = 256 always.
// Block: 256 thr = 8 waves; block tile 64(M)x128(N); wave tile 32x32 (2x2).
// LDS: 2 buffers x (A 64x32 hi+lo = 8KB, W 128x32 hi+lo = 16KB) = 48KB.
// Per K-step each thread issues 6 global_load_async_to_lds_b128; fragments
// are then read from LDS (ds_load_b128), deduped across the 8 waves.
// ---------------------------------------------------------------------------
template<bool HAS_PE, bool SPLIT_OUT>
__global__ void __launch_bounds__(256)
gemm_bf16split(const unsigned short* __restrict__ Ah, const unsigned short* __restrict__ Al,
               const unsigned short* __restrict__ Wh, const unsigned short* __restrict__ Wl,
               const float* __restrict__ bias,
               float* __restrict__ Cf, unsigned short* __restrict__ Ch,
               unsigned short* __restrict__ Cl,
               int N, int K, long aStride, long wStride, long cStride)
{
  __shared__ unsigned short sA[2][2][64 * 32];    // [buf][hi/lo][row*32+k]
  __shared__ unsigned short sW[2][2][128 * 32];

  const int lane = threadIdx.x & 31;
  const int wid  = threadIdx.x >> 5;
  const int b    = blockIdx.z;
  const int mBlock = blockIdx.y * 64;
  const int nBlock = blockIdx.x * 128;
  const int m0l  = (wid >> 2) * 32;               // wave tile inside block
  const int n0l  = (wid & 3) * 32;

  const unsigned short* Abh = Ah + (long)b * aStride;
  const unsigned short* Abl = Al + (long)b * aStride;
  const unsigned short* Wbh = Wh + (long)b * wStride;
  const unsigned short* Wbl = Wl + (long)b * wStride;

  const int r    = lane & 15;
  const int half = lane >> 4;

  // Cooperative tile fetch: thread t covers row t/4, 8-elem chunk (t%4)*8.
  const int rowA = threadIdx.x >> 2;              // 0..63
  const int col  = (threadIdx.x & 3) * 8;         // 0,8,16,24

  auto issue_tiles = [&](int buf, int k) {
    const long aoff = (long)(mBlock + rowA) * K + k + col;
    async_copy16(lds_off(&sA[buf][0][rowA * 32 + col]), Abh + aoff);
    async_copy16(lds_off(&sA[buf][1][rowA * 32 + col]), Abl + aoff);
    #pragma unroll
    for (int ph = 0; ph < 2; ++ph) {
      const int rowW = ph * 64 + rowA;
      const long woff = (long)(nBlock + rowW) * K + k + col;
      async_copy16(lds_off(&sW[buf][0][rowW * 32 + col]), Wbh + woff);
      async_copy16(lds_off(&sW[buf][1][rowW * 32 + col]), Wbl + woff);
    }
  };

  v8f c00 = {}, c01 = {}, c10 = {}, c11 = {};

  const int nSteps = K >> 5;
  issue_tiles(0, 0);

  for (int i = 0; i < nSteps; ++i) {
    wait_async0();           // my async writes for step i are in LDS
    __syncthreads();         // everyone's writes done; step i-1 reads done
    if (i + 1 < nSteps) issue_tiles((i + 1) & 1, (i + 1) * 32);

    const unsigned short* tAh = sA[i & 1][0];
    const unsigned short* tAl = sA[i & 1][1];
    const unsigned short* tWh = sW[i & 1][0];
    const unsigned short* tWl = sW[i & 1][1];

    v16bf ah0 = fragA_lds(tAh, m0l + r,      half);
    v16bf al0 = fragA_lds(tAl, m0l + r,      half);
    v16bf ah1 = fragA_lds(tAh, m0l + 16 + r, half);
    v16bf al1 = fragA_lds(tAl, m0l + 16 + r, half);
    v16bf bh0 = fragB_lds(tWh, n0l + r,      half);
    v16bf bl0 = fragB_lds(tWl, n0l + r,      half);
    v16bf bh1 = fragB_lds(tWh, n0l + 16 + r, half);
    v16bf bl1 = fragB_lds(tWl, n0l + 16 + r, half);

    c00 = wmma_bf16(ah0, bh0, c00);
    c00 = wmma_bf16(ah0, bl0, c00);
    c00 = wmma_bf16(al0, bh0, c00);

    c01 = wmma_bf16(ah0, bh1, c01);
    c01 = wmma_bf16(ah0, bl1, c01);
    c01 = wmma_bf16(al0, bh1, c01);

    c10 = wmma_bf16(ah1, bh0, c10);
    c10 = wmma_bf16(ah1, bl0, c10);
    c10 = wmma_bf16(al1, bh0, c10);

    c11 = wmma_bf16(ah1, bh1, c11);
    c11 = wmma_bf16(ah1, bl1, c11);
    c11 = wmma_bf16(al1, bh1, c11);
  }

  // Epilogue.  C/D f32 layout: VGPR e -> row (e + 8*half), col = lane&15.
  const int colBase = nBlock + n0l + (lane & 15);
  const int rowBase = mBlock + m0l + (half << 3);

  #pragma unroll
  for (int mi = 0; mi < 2; ++mi) {
    #pragma unroll
    for (int ni = 0; ni < 2; ++ni) {
      v8f acc = (mi == 0) ? (ni == 0 ? c00 : c01)
                          : (ni == 0 ? c10 : c11);
      const int col = colBase + ni * 16;
      const float bv = bias[col];
      #pragma unroll
      for (int e = 0; e < 8; ++e) {
        const int row = rowBase + mi * 16 + e;
        float v = fmaxf(acc[e] + bv, 0.0f);
        if (HAS_PE) v += pe_val(row, col, N);
        const long idx = (long)b * cStride + (long)row * N + col;
        if (SPLIT_OUT) {
          unsigned short hh, ll;
          split_bf16(v, hh, ll);
          Ch[idx] = hh;
          Cl[idx] = ll;
        } else {
          Cf[idx] = v;
        }
      }
    }
  }
}

// f32 tensor -> (hi, lo) bf16 split.
__global__ void split_kernel(const float* __restrict__ src,
                             unsigned short* __restrict__ hi,
                             unsigned short* __restrict__ lo, long n)
{
  long i = (long)blockIdx.x * blockDim.x + threadIdx.x;
  if (i >= n) return;
  split_bf16(src[i], hi[i], lo[i]);
}

// sK[row] = sum_k (hi+lo)[row,k]   (rows = B_*S_)
__global__ void rowsum_split_kernel(const unsigned short* __restrict__ hi,
                                    const unsigned short* __restrict__ lo,
                                    float* __restrict__ s, int rows, int K)
{
  int rIdx = blockIdx.x * blockDim.x + threadIdx.x;
  if (rIdx >= rows) return;
  const unsigned short* ph = hi + (long)rIdx * K;
  const unsigned short* pl = lo + (long)rIdx * K;
  float acc = 0.0f;
  for (int k = 0; k < K; ++k) acc += bf16_to_f(ph[k]) + bf16_to_f(pl[k]);
  s[rIdx] = acc;
}

// weff[b,o,j] = sum_k sK[b,k] * W[o,k,j]; W (f32) streamed once for all 8
// batches (bandwidth-bound roofline stage).  Output split to (hi,lo) bf16.
__global__ void __launch_bounds__(256)
weff_split_kernel(const float* __restrict__ sK,   // [B_, S_]
                  const float* __restrict__ W,    // [O, S_, J] f32
                  unsigned short* __restrict__ weffH,
                  unsigned short* __restrict__ weffL,
                  int O, int J)
{
  __shared__ float ls[B_ * S_];                   // 8 KB
  for (int t = threadIdx.x; t < B_ * S_; t += blockDim.x) ls[t] = sK[t];
  __syncthreads();

  long idx = (long)blockIdx.x * blockDim.x + threadIdx.x;
  if (idx >= (long)O * J) return;
  int o = (int)(idx / J);
  int j = (int)(idx % J);

  float acc[B_];
  #pragma unroll
  for (int bb = 0; bb < B_; ++bb) acc[bb] = 0.0f;

  const float* wp = W + ((long)o * S_) * J + j;
  for (int k = 0; k < S_; ++k) {
    float wv = wp[(long)k * J];                   // coalesced across j
    #pragma unroll
    for (int bb = 0; bb < B_; ++bb) acc[bb] += ls[bb * S_ + k] * wv;
  }
  #pragma unroll
  for (int bb = 0; bb < B_; ++bb) {
    unsigned short hh, ll;
    split_bf16(acc[bb], hh, ll);
    long oi = ((long)bb * O + o) * J + j;
    weffH[oi] = hh;
    weffL[oi] = ll;
  }
}

// out[b,v] = relu(sum_s h5[b,s,v] * wfin[s] + bfin)
__global__ void __launch_bounds__(256)
final_kernel(const float* __restrict__ h5, const float* __restrict__ wfin,
             const float* __restrict__ bfin, float* __restrict__ out)
{
  int idx = blockIdx.x * blockDim.x + threadIdx.x;  // B_*V_
  int b = idx / V_;
  int v = idx % V_;
  const float* hp = h5 + (long)b * S_ * V_ + v;
  float acc = 0.0f;
  for (int s = 0; s < S_; ++s) acc += hp[(long)s * V_] * wfin[s];
  out[idx] = fmaxf(acc + bfin[0], 0.0f);
}

extern "C" void kernel_launch(void* const* d_in, const int* in_sizes, int n_in,
                              void* d_out, int out_size, void* d_ws, size_t ws_size,
                              hipStream_t stream) {
  const float* x      = (const float*)d_in[0];   // [B,S,V]
  const float* w_emb  = (const float*)d_in[1];   // [D, V]
  const float* b_emb  = (const float*)d_in[2];
  const float* w_red  = (const float*)d_in[3];   // [2D, S*D]
  const float* b_red  = (const float*)d_in[4];
  const float* w_red2 = (const float*)d_in[5];   // [5D, S*2D]
  const float* b_red2 = (const float*)d_in[6];
  const float* w_up1  = (const float*)d_in[7];   // [10D, 5D]
  const float* b_up1  = (const float*)d_in[8];
  const float* w_up2  = (const float*)d_in[9];   // [V, 10D]
  const float* b_up2  = (const float*)d_in[10];
  const float* w_fin  = (const float*)d_in[11];  // [1, S]
  const float* b_fin  = (const float*)d_in[12];
  float* out = (float*)d_out;                    // [B, V]

  // ------------------- workspace layout (bump, ~85 MB) --------------------
  typedef unsigned short us;
  us* p = (us*)d_ws;
  us* xh  = p; p += 8388608;      // B*S*V
  us* xl  = p; p += 8388608;
  float* h5 = (float*)d_ws;       // 33.5 MB f32, ALIASES xh+xl (x dead after S1)
  us* weh = p; p += 524288;       // D*V
  us* wel = p; p += 524288;
  us* u1h = p; p += 819200;       // 10D*5D
  us* u1l = p; p += 819200;
  us* u2h = p; p += 5242880;      // V*10D
  us* u2l = p; p += 5242880;
  us* h1h = p; p += 262144;       // B*S*D
  us* h1l = p; p += 262144;
  us* h2h = p; p += 524288;       // B*S*2D
  us* h2l = p; p += 524288;
  us* h3h = p; p += 1310720;      // B*S*5D
  us* h3l = p; p += 1310720;
  us* h4h = p; p += 2621440;      // B*S*10D
  us* h4l = p; p += 2621440;
  us* w2h = p; p += 262144;       // B*2D*D
  us* w2l = p; p += 262144;
  us* w3h = p; p += 1310720;      // B*5D*2D
  us* w3l = p; p += 1310720;
  float* sK = (float*)p;          // B*S f32 (reused for both rowsums)

  // ---- split static operands to (hi,lo) bf16 ----
  split_kernel<<<dim3(8388608/256), 256, 0, stream>>>(x,     xh,  xl,  8388608L);
  split_kernel<<<dim3(524288/256),  256, 0, stream>>>(w_emb, weh, wel, 524288L);
  split_kernel<<<dim3(819200/256),  256, 0, stream>>>(w_up1, u1h, u1l, 819200L);
  split_kernel<<<dim3(5242880/256), 256, 0, stream>>>(w_up2, u2h, u2l, 5242880L);

  // Stage 1: h1 = relu(x . w_emb^T + b_emb) + P1                 [B,S,D]
  gemm_bf16split<true, true><<<dim3(D_/128, S_/64, B_), 256, 0, stream>>>(
      xh, xl, weh, wel, b_emb, nullptr, h1h, h1l,
      D_, V_, (long)S_*V_, 0, (long)S_*D_);

  // Stage 2 prep
  rowsum_split_kernel<<<dim3(B_*S_/256), 256, 0, stream>>>(h1h, h1l, sK, B_*S_, D_);
  weff_split_kernel<<<dim3(2*D_*D_/256), 256, 0, stream>>>(sK, w_red, w2h, w2l, 2*D_, D_);

  // Stage 2: h2 = relu(h1 . weff2^T + b_red) + P2                [B,S,2D]
  gemm_bf16split<true, true><<<dim3(2*D_/128, S_/64, B_), 256, 0, stream>>>(
      h1h, h1l, w2h, w2l, b_red, nullptr, h2h, h2l,
      2*D_, D_, (long)S_*D_, (long)2*D_*D_, (long)S_*2*D_);

  // Stage 3 prep
  rowsum_split_kernel<<<dim3(B_*S_/256), 256, 0, stream>>>(h2h, h2l, sK, B_*S_, 2*D_);
  weff_split_kernel<<<dim3(5*D_*2*D_/256), 256, 0, stream>>>(sK, w_red2, w3h, w3l, 5*D_, 2*D_);

  // Stage 3: h3 = relu(h2 . weff3^T + b_red2) + P3               [B,S,5D]
  gemm_bf16split<true, true><<<dim3(5*D_/128, S_/64, B_), 256, 0, stream>>>(
      h2h, h2l, w3h, w3l, b_red2, nullptr, h3h, h3l,
      5*D_, 2*D_, (long)S_*2*D_, (long)5*D_*2*D_, (long)S_*5*D_);

  // Stage 4: h4 = relu(h3 . w_up1^T + b_up1)                     [B,S,10D]
  gemm_bf16split<false, true><<<dim3(10*D_/128, S_/64, B_), 256, 0, stream>>>(
      h3h, h3l, u1h, u1l, b_up1, nullptr, h4h, h4l,
      10*D_, 5*D_, (long)S_*5*D_, 0, (long)S_*10*D_);

  // Stage 5: h5 = relu(h4 . w_up2^T + b_up2)  [B,S,V] f32 — 21.5 GFLOP
  gemm_bf16split<false, false><<<dim3(V_/128, S_/64, B_), 256, 0, stream>>>(
      h4h, h4l, u2h, u2l, b_up2, h5, nullptr, nullptr,
      V_, 10*D_, (long)S_*10*D_, 0, (long)S_*V_);

  // Stage 6: out[b,v] = relu(sum_s h5[b,s,v]*w_fin[s] + b_fin)
  final_kernel<<<dim3(B_*V_/256), 256, 0, stream>>>(h5, w_fin, b_fin, out);
}
